// SelfAttention_63307817943251
// MI455X (gfx1250) — matrix operand order repaired and checked by reading, not compiled, not run on hardware
//
#include <hip/hip_runtime.h>

#define DM 6144
#define SEQ 2048
#define NH 64
#define DH 96
#define BK 64

typedef __bf16 bf16_t;
typedef bf16_t v16bf __attribute__((ext_vector_type(16)));
typedef bf16_t v2bf __attribute__((ext_vector_type(2)));
typedef float v8f __attribute__((ext_vector_type(8)));

#if defined(__HIP_DEVICE_COMPILE__) && __has_builtin(__builtin_amdgcn_global_load_async_to_lds_b128) && __has_builtin(__builtin_amdgcn_s_wait_asynccnt)
#define USE_ASYNC_LDS 1
#else
#define USE_ASYNC_LDS 0
#endif

#if USE_ASYNC_LDS
typedef int v4i_gcc __attribute__((vector_size(16)));
typedef __attribute__((address_space(1))) v4i_gcc* as1_v4i;
typedef __attribute__((address_space(3))) v4i_gcc* as3_v4i;

// 16-byte async global -> LDS copy (tracked by ASYNCcnt, no VGPR round-trip)
__device__ inline void async_copy_b128(const void* g, void* l) {
    __builtin_amdgcn_global_load_async_to_lds_b128((as1_v4i)g, (as3_v4i)l, 0, 0);
}
#endif

union FragA {
    v16bf v;
    uint4 u[2];
};

// fp32 -> bf16 (RNE).  Prefer the hardware pack-convert when available.
#if defined(__HIP_DEVICE_COMPILE__) && __has_builtin(__builtin_amdgcn_cvt_pk_bf16_f32)
__device__ inline unsigned int f2bf2(float lo, float hi) {
    v2bf r = __builtin_amdgcn_cvt_pk_bf16_f32(lo, hi);
    return __builtin_bit_cast(unsigned int, r);
}
__device__ inline unsigned short f2bf(float x) {
    return (unsigned short)(f2bf2(x, 0.0f) & 0xFFFFu);
}
#else
__device__ inline unsigned short f2bf(float x) {
    unsigned int u = __float_as_uint(x);
    u += 0x7FFFu + ((u >> 16) & 1u);   // round-to-nearest-even
    return (unsigned short)(u >> 16);
}
__device__ inline unsigned int f2bf2(float lo, float hi) {
    return (unsigned int)f2bf(lo) | ((unsigned int)f2bf(hi) << 16);
}
#endif

__device__ inline v8f wmma_bf16(const FragA& a, const FragA& b, v8f c) {
    return __builtin_amdgcn_wmma_f32_16x16x32_bf16(
        /*neg_a=*/false, a.v, /*neg_b=*/false, b.v,
        /*c_mod=*/(short)0, c, /*reuse_a=*/false, /*reuse_b=*/false);
}

// ---------------------------------------------------------------------------
// Kernel 1: Y = X * W + bias  (fp32 in, bf16 out), fused over Q/K/V via grid.z
// Block tile 128x128, K-step 64.  8 waves, each wave computes 64x32.
// Grid: (M/128, N/128, 3) -> consecutive blocks share the same W strip (L2).
// ---------------------------------------------------------------------------
__global__ __launch_bounds__(256) void qkv_gemm_kernel(
    const float* __restrict__ X,
    const float* __restrict__ Wq, const float* __restrict__ Wk, const float* __restrict__ Wv,
    const float* __restrict__ bq, const float* __restrict__ bk, const float* __restrict__ bv,
    unsigned short* __restrict__ Qb, unsigned short* __restrict__ Kb, unsigned short* __restrict__ Vb)
{
    const int z = blockIdx.z;
    const float* __restrict__ W    = (z == 0) ? Wq : (z == 1) ? Wk : Wv;
    const float* __restrict__ bias = (z == 0) ? bq : (z == 1) ? bk : bv;
    unsigned short* __restrict__ out = (z == 0) ? Qb : (z == 1) ? Kb : Vb;
    const float scale = (z == 0) ? 0.10206207261596575f : 1.0f; // fold 1/sqrt(96) into Q

    const int m0   = blockIdx.x * 128;   // M fastest: W strip reuse in L2
    const int n0   = blockIdx.y * 128;
    const int tid  = threadIdx.x;
    const int wave = tid >> 5;
    const int lane = tid & 31;
    const int half = lane >> 4;
    const int l16  = lane & 15;
    const int waveM = wave >> 2;  // 0..1 : 64 rows each
    const int waveN = wave & 3;   // 0..3 : 32 cols each

    __shared__ __align__(16) unsigned short lA[128 * BK];   // [m][k] bf16, 16 KB
    __shared__ __align__(16) unsigned short lBt[128 * BK];  // [n][k] bf16, 16 KB

    v8f acc[4][2];
#pragma unroll
    for (int i = 0; i < 4; i++)
#pragma unroll
        for (int j = 0; j < 2; j++) acc[i][j] = v8f{};

    const int arow = tid >> 1;    // 0..127
    const int aseg = tid & 1;     // 32 floats along k
    const int bkp  = tid >> 3;    // 0..31 : k-pair (k = 2*bkp, 2*bkp+1)
    const int bng  = tid & 7;     // 16 floats along n

    for (int k0 = 0; k0 < DM; k0 += BK) {
        // Stage A tile (128x64) fp32 -> packed bf16, vector LDS stores
        {
            const float4* src = (const float4*)(X + (size_t)(m0 + arow) * DM + k0 + aseg * 32);
            uint4* dst = (uint4*)&lA[arow * BK + aseg * 32];
#pragma unroll
            for (int i = 0; i < 4; i++) {
                float4 f0 = src[2 * i];
                float4 f1 = src[2 * i + 1];
                uint4 p;
                p.x = f2bf2(f0.x, f0.y);
                p.y = f2bf2(f0.z, f0.w);
                p.z = f2bf2(f1.x, f1.y);
                p.w = f2bf2(f1.z, f1.w);
                dst[i] = p;
            }
        }
        // Stage B tile (64x128) fp32 -> transposed [n][k], packed k-pairs (b32 stores)
        {
            const float4* r0 = (const float4*)(W + (size_t)(k0 + 2 * bkp) * DM + n0 + bng * 16);
            const float4* r1 = (const float4*)((const float*)r0 + DM);
#pragma unroll
            for (int i = 0; i < 4; i++) {
                float4 a = r0[i];
                float4 b = r1[i];
                int nb = bng * 16 + i * 4;
                *(unsigned int*)&lBt[(nb + 0) * BK + 2 * bkp] = f2bf2(a.x, b.x);
                *(unsigned int*)&lBt[(nb + 1) * BK + 2 * bkp] = f2bf2(a.y, b.y);
                *(unsigned int*)&lBt[(nb + 2) * BK + 2 * bkp] = f2bf2(a.z, b.z);
                *(unsigned int*)&lBt[(nb + 3) * BK + 2 * bkp] = f2bf2(a.w, b.w);
            }
        }
        __syncthreads();

#pragma unroll
        for (int ks = 0; ks < 2; ks++) {
            FragA afrag[4], bfrag[2];
#pragma unroll
            for (int mt = 0; mt < 4; mt++) {
                const unsigned short* p = &lA[(waveM * 64 + mt * 16 + l16) * BK + ks * 32];
                afrag[mt].u[0] = *(const uint4*)(p + 16 * half);
                afrag[mt].u[1] = *(const uint4*)(p + 16 * half + 8);
            }
#pragma unroll
            for (int nt = 0; nt < 2; nt++) {
                const unsigned short* p = &lBt[(waveN * 32 + nt * 16 + l16) * BK + ks * 32 + 16 * half];
                bfrag[nt].u[0] = *(const uint4*)(p);
                bfrag[nt].u[1] = *(const uint4*)(p + 8);
            }
#pragma unroll
            for (int mt = 0; mt < 4; mt++)
#pragma unroll
                for (int nt = 0; nt < 2; nt++)
                    acc[mt][nt] = wmma_bf16(afrag[mt], bfrag[nt], acc[mt][nt]);
        }
        __syncthreads();
    }

    // Epilogue: bias add, scale, bf16 store
#pragma unroll
    for (int nt = 0; nt < 2; nt++) {
        const int col = n0 + waveN * 32 + nt * 16 + l16;
        const float bcol = bias[col];
#pragma unroll
        for (int mt = 0; mt < 4; mt++) {
#pragma unroll
            for (int j = 0; j < 8; j++) {
                int row = m0 + waveM * 64 + mt * 16 + j + 8 * half;
                float v = (acc[mt][nt][j] + bcol) * scale;
                out[(size_t)row * DM + col] = f2bf(v);
            }
        }
    }
}

// ---------------------------------------------------------------------------
// Kernel 2: flash attention.  grid (SEQ/128, NH), 8 waves per block, each wave
// owns 16 query rows.  Loop over keys in chunks of 32 with online softmax.
// ---------------------------------------------------------------------------
__global__ __launch_bounds__(256) void flash_attn_kernel(
    const unsigned short* __restrict__ Qb,
    const unsigned short* __restrict__ Kb,
    const unsigned short* __restrict__ Vb,
    float* __restrict__ out)
{
    const int h    = blockIdx.y;
    const int tid  = threadIdx.x;
    const int wave = tid >> 5;
    const int lane = tid & 31;
    const int half = lane >> 4;
    const int l16  = lane & 15;
    const int q0   = blockIdx.x * 128 + wave * 16;

    __shared__ __align__(16) unsigned short lK[32 * 96];     // [key][d]
    __shared__ __align__(16) unsigned short lVt[96 * 32];    // [d][key]
    __shared__ __align__(16) unsigned short lP[8][16 * 32];  // per-wave P patch

    // Q fragments: A-matrix layout, 3 d-chunks of 32 (DH = 96), straight from global.
    FragA qf[3];
    {
        const unsigned short* qrow = Qb + (size_t)(q0 + l16) * DM + h * DH;
#pragma unroll
        for (int dc = 0; dc < 3; dc++) {
            qf[dc].u[0] = *(const uint4*)(qrow + dc * 32 + 16 * half);
            qf[dc].u[1] = *(const uint4*)(qrow + dc * 32 + 16 * half + 8);
        }
    }

    v8f o[6];
#pragma unroll
    for (int i = 0; i < 6; i++) o[i] = v8f{};
    float mrow[8], lrow[8];
#pragma unroll
    for (int j = 0; j < 8; j++) { mrow[j] = -3.0e38f; lrow[j] = 0.0f; }

    // V-transpose staging: thread -> key pair (2*vkp, 2*vkp+1), 6 dims
    const int vkp = tid >> 4;          // 0..15
    const int vds = (tid & 15) * 6;    // 0,6,...,90
#if !USE_ASYNC_LDS
    const int skey = tid >> 3;         // 0..31
    const int d0   = (tid & 7) * 12;   // 12 dims per thread
#endif

    for (int key0 = 0; key0 < SEQ; key0 += 32) {
        // ---- Stage K [key][d] ----
#if USE_ASYNC_LDS
        {
            // 32 keys x 96 dims = 384 16-byte chunks; chunk c: key=c/12, seg=c%12
            int key = tid / 12, seg = tid % 12;
            async_copy_b128(Kb + (size_t)(key0 + key) * DM + h * DH + seg * 8,
                            &lK[key * 96 + seg * 8]);
            if (tid < 128) {
                int c1 = tid + 256;
                int key1 = c1 / 12, seg1 = c1 % 12;
                async_copy_b128(Kb + (size_t)(key0 + key1) * DM + h * DH + seg1 * 8,
                                &lK[key1 * 96 + seg1 * 8]);
            }
            __builtin_amdgcn_s_wait_asynccnt(0);
        }
#else
        {
            const unsigned int* k32 =
                (const unsigned int*)(Kb + (size_t)(key0 + skey) * DM + h * DH + d0);
            unsigned int* kd = (unsigned int*)&lK[skey * 96 + d0];
#pragma unroll
            for (int i = 0; i < 6; i++) kd[i] = k32[i];
        }
#endif
        // ---- Stage V transposed [d][key], packed key-pairs ----
        {
            const unsigned int* ga =
                (const unsigned int*)(Vb + (size_t)(key0 + 2 * vkp) * DM + h * DH + vds);
            const unsigned int* gb = (const unsigned int*)((const unsigned short*)ga + DM);
            unsigned int a0 = ga[0], a1 = ga[1], a2 = ga[2];
            unsigned int b0 = gb[0], b1 = gb[1], b2 = gb[2];
            unsigned int* vt = (unsigned int*)lVt;
            vt[(vds + 0) * 16 + vkp] = (a0 & 0xFFFFu) | (b0 << 16);
            vt[(vds + 1) * 16 + vkp] = (a0 >> 16) | (b0 & 0xFFFF0000u);
            vt[(vds + 2) * 16 + vkp] = (a1 & 0xFFFFu) | (b1 << 16);
            vt[(vds + 3) * 16 + vkp] = (a1 >> 16) | (b1 & 0xFFFF0000u);
            vt[(vds + 4) * 16 + vkp] = (a2 & 0xFFFFu) | (b2 << 16);
            vt[(vds + 5) * 16 + vkp] = (a2 >> 16) | (b2 & 0xFFFF0000u);
        }
        __syncthreads();

        // S = Q * K^T : two 16-key subtiles, 3 WMMAs each over d
        v8f s[2];
#pragma unroll
        for (int sub = 0; sub < 2; sub++) {
            v8f st = v8f{};
#pragma unroll
            for (int dc = 0; dc < 3; dc++) {
                FragA kf;
                const unsigned short* p = &lK[(sub * 16 + l16) * 96 + dc * 32 + 16 * half];
                kf.u[0] = *(const uint4*)(p);
                kf.u[1] = *(const uint4*)(p + 8);
                st = wmma_bf16(qf[dc], kf, st);
            }
            s[sub] = st;
        }

        // Online softmax per row (row = j + 8*half; 16-lane half-group reduction)
#pragma unroll
        for (int j = 0; j < 8; j++) {
            float mx = fmaxf(s[0][j], s[1][j]);
            mx = fmaxf(mx, __shfl_xor(mx, 1, 32));
            mx = fmaxf(mx, __shfl_xor(mx, 2, 32));
            mx = fmaxf(mx, __shfl_xor(mx, 4, 32));
            mx = fmaxf(mx, __shfl_xor(mx, 8, 32));
            float mi = fmaxf(mrow[j], mx);
            float alpha = __expf(mrow[j] - mi);
            mrow[j] = mi;
            float p0 = __expf(s[0][j] - mi);
            float p1 = __expf(s[1][j] - mi);
            float rs = p0 + p1;
            rs += __shfl_xor(rs, 1, 32);
            rs += __shfl_xor(rs, 2, 32);
            rs += __shfl_xor(rs, 4, 32);
            rs += __shfl_xor(rs, 8, 32);
            lrow[j] = lrow[j] * alpha + rs;
#pragma unroll
            for (int n = 0; n < 6; n++) o[n][j] *= alpha;
            int prow = j + 8 * half;
            lP[wave][prow * 32 + l16]      = f2bf(p0);
            lP[wave][prow * 32 + 16 + l16] = f2bf(p1);
        }

        // O += P (16x32) * V (32x96) : reload P in A-layout from per-wave LDS patch
        FragA pf;
        {
            const unsigned short* p = &lP[wave][l16 * 32];
            pf.u[0] = *(const uint4*)(p + 16 * half);
            pf.u[1] = *(const uint4*)(p + 16 * half + 8);
        }
#pragma unroll
        for (int nt = 0; nt < 6; nt++) {
            FragA vf;
            const unsigned short* p = &lVt[(nt * 16 + l16) * 32 + 16 * half];
            vf.u[0] = *(const uint4*)(p);
            vf.u[1] = *(const uint4*)(p + 8);
            o[nt] = wmma_bf16(pf, vf, o[nt]);
        }
        __syncthreads();
    }

    // Normalize and store fp32 context [S, H*DH]
#pragma unroll
    for (int j = 0; j < 8; j++) {
        float inv = 1.0f / lrow[j];
        int row = q0 + j + 8 * half;
#pragma unroll
        for (int nt = 0; nt < 6; nt++) {
            out[(size_t)row * DM + h * DH + nt * 16 + l16] = o[nt][j] * inv;
        }
    }
}

extern "C" void kernel_launch(void* const* d_in, const int* in_sizes, int n_in,
                              void* d_out, int out_size, void* d_ws, size_t ws_size,
                              hipStream_t stream) {
    const float* X  = (const float*)d_in[0];
    const float* Wq = (const float*)d_in[1];
    const float* bq = (const float*)d_in[2];
    const float* Wk = (const float*)d_in[3];
    const float* bk = (const float*)d_in[4];
    const float* Wv = (const float*)d_in[5];
    const float* bv = (const float*)d_in[6];
    float* out = (float*)d_out;

    const size_t bufElems = (size_t)SEQ * DM;          // bf16 elements per buffer
    unsigned short* Qb = (unsigned short*)d_ws;        // 25.2 MB
    unsigned short* Kb = Qb + bufElems;                // 25.2 MB
    unsigned short* Vb = Kb + bufElems;                // 25.2 MB

    qkv_gemm_kernel<<<dim3(SEQ / 128, DM / 128, 3), 256, 0, stream>>>(
        X, Wq, Wk, Wv, bq, bk, bv, Qb, Kb, Vb);
    flash_attn_kernel<<<dim3(SEQ / 128, NH), 256, 0, stream>>>(Qb, Kb, Vb, out);
}
